// EpisodicBuffer_39737037422860
// MI455X (gfx1250) — matrix operand order repaired and checked by reading, not compiled, not run on hardware
//
#include <hip/hip_runtime.h>

#define DEVI __device__ __forceinline__

typedef __attribute__((ext_vector_type(16))) __bf16 v16bf;
typedef __attribute__((ext_vector_type(8)))  float  v8f;

union AF { v16bf v; uint4 u[2]; };

constexpr int BS = 4096;   // B*S
constexpr int H  = 256;
constexpr int M  = 32768;
constexpr int E  = 256;
constexpr int V  = 32000;
constexpr int F1 = 512;
constexpr int MSPLIT = 8;
constexpr int MCHUNK = M / MSPLIT;     // 4096 contexts per split
constexpr int NSTEP  = MCHUNK / 32;    // 128

DEVI v8f wmma_bf16(v16bf a, v16bf b, v8f c) {
  return __builtin_amdgcn_wmma_f32_16x16x32_bf16(false, a, false, b, (short)0, c,
                                                 false, false);
}

// --- CDNA5 async global->LDS copy (16B per lane), tracked by ASYNCcnt --------
DEVI void async_b128(void* lds_dst, const void* gsrc) {
  unsigned lds_off = (unsigned)(size_t)lds_dst;  // low 32 bits = LDS byte addr
  asm volatile("global_load_async_to_lds_b128 %0, %1, off"
               :: "v"(lds_off), "v"(gsrc) : "memory");
}
DEVI void async_wait() {
  asm volatile("s_wait_asynccnt 0x0" ::: "memory");
}

// ---------------------------------------------------------------- normalize
__global__ void k_norm_rows(const float* __restrict__ x, __bf16* __restrict__ out) {
  __shared__ float red[256];
  const int row = blockIdx.x, t = threadIdx.x;
  float v = x[row * 256 + t];
  red[t] = v * v;
  __syncthreads();
  #pragma unroll
  for (int s = 128; s > 0; s >>= 1) {
    if (t < s) red[t] += red[t + s];
    __syncthreads();
  }
  const float inv = 1.0f / fmaxf(sqrtf(red[0]), 1e-8f);
  out[row * 256 + t] = (__bf16)(v * inv);
}

// ---------------------------------------------------------------- f32 -> bf16
__global__ void k_cvt_bf16(const float* __restrict__ x, __bf16* __restrict__ out, int n) {
  const int i = blockIdx.x * 256 + threadIdx.x;
  if (i < n) out[i] = (__bf16)x[i];
}

// ------------------------------------------------- transpose embeddeds -> bf16
__global__ void k_transpose_emb(const float* __restrict__ emb, __bf16* __restrict__ eT) {
  __shared__ float tile[32][33];
  const int b = blockIdx.x;
  const int mt = b % (M / 32), et = b / (M / 32);
  const int tx = threadIdx.x & 31, ty = threadIdx.x >> 5;
  #pragma unroll
  for (int i = 0; i < 32; i += 8)
    tile[ty + i][tx] = emb[(mt * 32 + ty + i) * E + et * 32 + tx];
  __syncthreads();
  #pragma unroll
  for (int i = 0; i < 32; i += 8)
    eT[(et * 32 + ty + i) * M + mt * 32 + tx] = (__bf16)tile[tx][ty + i];
}

// ------------------------------------------------- fused sims/softmax/retrieve
__global__ void __launch_bounds__(256, 1) k_retrieval_partial(
    const __bf16* __restrict__ qn, const __bf16* __restrict__ cn,
    const __bf16* __restrict__ eT, float* __restrict__ pacc,
    float* __restrict__ pmax, float* __restrict__ psum) {
  __shared__ __align__(16) __bf16 l_cn[32][256];      // [ctx][h]     16KB
  __shared__ __align__(16) __bf16 l_e[2][256][32];    // [buf][e][ctx] 32KB
  __shared__ __align__(16) __bf16 s_w[8][16][32];     // per-wave weights 8KB

  const int tid  = threadIdx.x;
  const int wave = tid >> 5, lane = tid & 31;
  const int hi = lane >> 4, lo = lane & 15;
  const int qg = blockIdx.x >> 3;          // 0..31
  const int split = blockIdx.x & 7;
  const int qt = qg * 8 + wave;            // 0..255
  const int q0 = qt * 16;
  const int mbase = split * MCHUNK;

  // staging geometry (per thread)
  const int srow = tid >> 3, sch = tid & 7;       // cn: 32 rows x 512B

  // 8 A fragments: qn[16 x 256]
  AF a[8];
  #pragma unroll
  for (int kk = 0; kk < 8; kk++) {
    const __bf16* p = qn + (q0 + lo) * H + kk * 32 + hi * 16;
    a[kk].u[0] = *(const uint4*)p;
    a[kk].u[1] = *(const uint4*)(p + 8);
  }

  v8f acc[16];
  #pragma unroll
  for (int t = 0; t < 16; t++) acc[t] = (v8f)0.0f;

  float rmax[8], rsum[8];                  // rows r+hi*8, replicated over lo
  #pragma unroll
  for (int r = 0; r < 8; r++) { rmax[r] = -3.0e38f; rsum[r] = 0.0f; }

  // prologue staging
  {
    const __bf16* src = cn + (size_t)(mbase + srow) * H + sch * 32;
    __bf16* dst = &l_cn[srow][sch * 32];
    #pragma unroll
    for (int j = 0; j < 4; j++) async_b128(dst + j * 8, src + j * 8);
    const __bf16* se = eT + (size_t)tid * M + mbase;
    __bf16* de = &l_e[0][tid][0];
    #pragma unroll
    for (int j = 0; j < 4; j++) async_b128(de + j * 8, se + j * 8);
  }
  async_wait();
  __syncthreads();

  int buf = 0;
  for (int step = 0; step < NSTEP; step++) {
    const int mt = mbase + step * 32;
    const bool last = (step == NSTEP - 1);

    // stage next eT tile into the other buffer (overlaps whole iteration)
    if (!last) {
      const __bf16* se = eT + (size_t)tid * M + mt + 32;
      __bf16* de = &l_e[buf ^ 1][tid][0];
      #pragma unroll
      for (int j = 0; j < 4; j++) async_b128(de + j * 8, se + j * 8);
    }

    // ---- sims: two 16x16 C fragments over 32 contexts
    v8f s0 = (v8f)0.0f, s1 = (v8f)0.0f;
    #pragma unroll
    for (int kk = 0; kk < 8; kk++) {
      AF b0, b1;
      const __bf16* pb0 = &l_cn[lo][kk * 32 + hi * 16];
      const __bf16* pb1 = &l_cn[16 + lo][kk * 32 + hi * 16];
      b0.u[0] = *(const uint4*)pb0; b0.u[1] = *(const uint4*)(pb0 + 8);
      b1.u[0] = *(const uint4*)pb1; b1.u[1] = *(const uint4*)(pb1 + 8);
      s0 = wmma_bf16(a[kk].v, b0.v, s0);
      s1 = wmma_bf16(a[kk].v, b1.v, s1);
    }
    __syncthreads();                        // all waves done reading l_cn

    // restage cn tile for next step (overlaps softmax + retrieve)
    if (!last) {
      const __bf16* src = cn + (size_t)(mt + 32 + srow) * H + sch * 32;
      __bf16* dst = &l_cn[srow][sch * 32];
      #pragma unroll
      for (int j = 0; j < 4; j++) async_b128(dst + j * 8, src + j * 8);
    }

    // ---- online softmax in registers (rows r+hi*8; 16-lane shuffles)
    float scale[8];
    bool chg = false;
    #pragma unroll
    for (int r = 0; r < 8; r++) {
      float tm = fmaxf(s0[r], s1[r]);
      tm = fmaxf(tm, __shfl_xor(tm, 1));
      tm = fmaxf(tm, __shfl_xor(tm, 2));
      tm = fmaxf(tm, __shfl_xor(tm, 4));
      tm = fmaxf(tm, __shfl_xor(tm, 8));
      const float nm = fmaxf(rmax[r], tm);
      const float sc = __expf(100.0f * (rmax[r] - nm));
      const float e0 = __expf(100.0f * (s0[r] - nm));
      const float e1 = __expf(100.0f * (s1[r] - nm));
      float ts = e0 + e1;
      ts += __shfl_xor(ts, 1);
      ts += __shfl_xor(ts, 2);
      ts += __shfl_xor(ts, 4);
      ts += __shfl_xor(ts, 8);
      rsum[r] = rsum[r] * sc + ts;
      rmax[r] = nm;
      scale[r] = sc;
      chg |= (sc != 1.0f);
      s_w[wave][r + hi * 8][lo]      = (__bf16)e0;
      s_w[wave][r + hi * 8][lo + 16] = (__bf16)e1;
    }
    __builtin_amdgcn_wave_barrier();

    if (__any(chg)) {
      #pragma unroll
      for (int t = 0; t < 16; t++)
        #pragma unroll
        for (int r = 0; r < 8; r++) acc[t][r] *= scale[r];
    }

    // ---- A fragment from weights tile (16x32, K = contexts)
    AF aw;
    {
      const __bf16* p = &s_w[wave][lo][hi * 16];
      aw.u[0] = *(const uint4*)p;
      aw.u[1] = *(const uint4*)(p + 8);
    }

    // ---- retrieved += W(16x32) * E(32x256); 2-deep pipelined B loads
    AF bb[2];
    {
      const __bf16* pb = &l_e[buf][lo][hi * 16];
      bb[0].u[0] = *(const uint4*)pb; bb[0].u[1] = *(const uint4*)(pb + 8);
    }
    #pragma unroll
    for (int t = 0; t < 16; t++) {
      if (t < 15) {
        const __bf16* pb = &l_e[buf][(t + 1) * 16 + lo][hi * 16];
        bb[(t + 1) & 1].u[0] = *(const uint4*)pb;
        bb[(t + 1) & 1].u[1] = *(const uint4*)(pb + 8);
      }
      acc[t] = wmma_bf16(aw.v, bb[t & 1].v, acc[t]);
    }

    async_wait();        // next-step staging has landed
    __syncthreads();
    buf ^= 1;
  }

  // ---- write partial accumulator + stats (z-space = sims * 100)
  float* pa = pacc + (size_t)(qt * MSPLIT + split) * 16 * E;
  #pragma unroll
  for (int t = 0; t < 16; t++)
    #pragma unroll
    for (int r = 0; r < 8; r++)
      pa[(r + hi * 8) * E + t * 16 + lo] = acc[t][r];
  if (lo == 0) {
    const int base = (qt * MSPLIT + split) * 16;
    #pragma unroll
    for (int r = 0; r < 8; r++) {
      pmax[base + r + hi * 8] = rmax[r] * 100.0f;
      psum[base + r + hi * 8] = rsum[r];
    }
  }
}

// ------------------------------------------------- merge M-split partials
__global__ void k_combine(const float* __restrict__ pacc, const float* __restrict__ pmax,
                          const float* __restrict__ psum, __bf16* __restrict__ retr) {
  const int idx = blockIdx.x * blockDim.x + threadIdx.x;  // over BS*E
  const int q = idx >> 8, e = idx & 255;
  const int qt = q >> 4, qr = q & 15;
  float gmax = -3.0e38f;
  #pragma unroll
  for (int s = 0; s < MSPLIT; s++)
    gmax = fmaxf(gmax, pmax[(qt * MSPLIT + s) * 16 + qr]);
  float L = 0.f, acc = 0.f;
  #pragma unroll
  for (int s = 0; s < MSPLIT; s++) {
    const int base = (qt * MSPLIT + s) * 16 + qr;
    const float f = __expf(pmax[base] - gmax);
    L += psum[base] * f;
    acc += pacc[(size_t)base * E + e] * f;
  }
  retr[q * E + e] = (__bf16)(acc / L);
}

// ------------------------------------------------- h = relu(retr @ fc1_w^T + b)
__global__ void __launch_bounds__(256) k_fc1(const __bf16* __restrict__ retr,
    const __bf16* __restrict__ w, const float* __restrict__ bias,
    __bf16* __restrict__ h) {
  const int wave = threadIdx.x >> 5, lane = threadIdx.x & 31;
  const int hi = lane >> 4, lo = lane & 15;
  const int tile = blockIdx.x * 8 + wave;          // (BS/16)*(F1/16) = 8192
  const int qt = tile >> 5, nt = tile & 31;
  const int n = nt * 16 + lo;
  v8f c = (v8f)0.0f;
  #pragma unroll
  for (int kk = 0; kk < E / 32; kk++) {
    AF a, b;
    const __bf16* pa = retr + (qt * 16 + lo) * E + kk * 32 + hi * 16;
    a.u[0] = *(const uint4*)pa; a.u[1] = *(const uint4*)(pa + 8);
    const __bf16* pb = w + (size_t)n * E + kk * 32 + hi * 16;
    b.u[0] = *(const uint4*)pb; b.u[1] = *(const uint4*)(pb + 8);
    c = wmma_bf16(a.v, b.v, c);
  }
  const float bn = bias[n];
  #pragma unroll
  for (int r = 0; r < 8; r++) {
    const float v = fmaxf(c[r] + bn, 0.0f);
    h[(qt * 16 + r + hi * 8) * F1 + n] = (__bf16)v;
  }
}

// ------------------------------------------------- out = h @ fc2_w^T + b
// LDS-blocked 128q x 128v; double-buffered async B staging, pipelined frags.
__global__ void __launch_bounds__(256, 1) k_fc2(const __bf16* __restrict__ h,
    const __bf16* __restrict__ w, const float* __restrict__ bias,
    float* __restrict__ out) {
  __shared__ __align__(16) __bf16 lB[2][128][32];    // 16KB
  const int tid = threadIdx.x;
  const int wave = tid >> 5, lane = tid & 31;
  const int hi = lane >> 4, lo = lane & 15;
  const int qb = blockIdx.x & 31;                 // 32 q-blocks of 128
  const int vb = blockIdx.x >> 5;                 // 250 v-blocks of 128
  const int q0 = qb * 128 + wave * 16;
  const int sv = tid >> 1, shf = tid & 1;         // staging: half-row per thread

  AF a[16];
  #pragma unroll
  for (int kk = 0; kk < 16; kk++) {
    const __bf16* pa = h + (size_t)(q0 + lo) * F1 + kk * 32 + hi * 16;
    a[kk].u[0] = *(const uint4*)pa;
    a[kk].u[1] = *(const uint4*)(pa + 8);
  }
  v8f c[8];
  #pragma unroll
  for (int vt = 0; vt < 8; vt++) c[vt] = (v8f)0.0f;

  // prologue stage kk=0
  {
    const __bf16* src = w + (size_t)(vb * 128 + sv) * F1 + shf * 16;
    __bf16* dst = &lB[0][sv][shf * 16];
    async_b128(dst, src);
    async_b128(dst + 8, src + 8);
  }
  async_wait();
  __syncthreads();

  for (int kk = 0; kk < 16; kk++) {
    const int cur = kk & 1;
    if (kk < 15) {   // stage next K-slice into other buffer (overlaps compute)
      const __bf16* src = w + (size_t)(vb * 128 + sv) * F1 + (kk + 1) * 32 + shf * 16;
      __bf16* dst = &lB[cur ^ 1][sv][shf * 16];
      async_b128(dst, src);
      async_b128(dst + 8, src + 8);
    }
    AF bb[2];
    {
      const __bf16* pb = &lB[cur][lo][hi * 16];
      bb[0].u[0] = *(const uint4*)pb; bb[0].u[1] = *(const uint4*)(pb + 8);
    }
    #pragma unroll
    for (int vt = 0; vt < 8; vt++) {
      if (vt < 7) {
        const __bf16* pb = &lB[cur][(vt + 1) * 16 + lo][hi * 16];
        bb[(vt + 1) & 1].u[0] = *(const uint4*)pb;
        bb[(vt + 1) & 1].u[1] = *(const uint4*)(pb + 8);
      }
      c[vt] = wmma_bf16(a[kk].v, bb[vt & 1].v, c[vt]);
    }
    async_wait();
    __syncthreads();
  }

  #pragma unroll
  for (int vt = 0; vt < 8; vt++) {
    const int n = vb * 128 + vt * 16 + lo;
    const float bn = bias[n];
    #pragma unroll
    for (int r = 0; r < 8; r++)
      out[(size_t)(q0 + r + hi * 8) * V + n] = c[vt][r] + bn;
  }
}

// ----------------------------------------------------------------- launcher
extern "C" void kernel_launch(void* const* d_in, const int* in_sizes, int n_in,
                              void* d_out, int out_size, void* d_ws, size_t ws_size,
                              hipStream_t stream) {
  const float* query     = (const float*)d_in[0];   // [8,512,256]
  const float* contexts  = (const float*)d_in[1];   // [32768,256]
  const float* embeddeds = (const float*)d_in[2];   // [32768,256]
  const float* fc1_w     = (const float*)d_in[3];   // [512,256]
  const float* fc1_b     = (const float*)d_in[4];   // [512]
  const float* fc2_w     = (const float*)d_in[5];   // [32000,512]
  const float* fc2_b     = (const float*)d_in[6];   // [32000]

  char* ws = (char*)d_ws;
  size_t off = 0;
  __bf16* qn    = (__bf16*)(ws + off); off += (size_t)BS * H * 2;
  __bf16* cnb   = (__bf16*)(ws + off); off += (size_t)M * H * 2;
  __bf16* eT    = (__bf16*)(ws + off); off += (size_t)E * M * 2;
  float*  pacc  = (float*) (ws + off); off += (size_t)(BS/16) * MSPLIT * 16 * E * 4;
  float*  pmax  = (float*) (ws + off); off += (size_t)(BS/16) * MSPLIT * 16 * 4;
  float*  psum  = (float*) (ws + off); off += (size_t)(BS/16) * MSPLIT * 16 * 4;
  __bf16* retr  = (__bf16*)(ws + off); off += (size_t)BS * E * 2;
  __bf16* hbuf  = (__bf16*)(ws + off); off += (size_t)BS * F1 * 2;
  __bf16* w1b   = (__bf16*)(ws + off); off += (size_t)F1 * E * 2;
  __bf16* w2b   = (__bf16*)(ws + off); off += (size_t)V * F1 * 2;

  k_norm_rows<<<BS, 256, 0, stream>>>(query, qn);
  k_norm_rows<<<M, 256, 0, stream>>>(contexts, cnb);
  k_transpose_emb<<<(M / 32) * (E / 32), 256, 0, stream>>>(embeddeds, eT);
  k_cvt_bf16<<<(F1 * E + 255) / 256, 256, 0, stream>>>(fc1_w, w1b, F1 * E);
  k_cvt_bf16<<<(V * F1 + 255) / 256, 256, 0, stream>>>(fc2_w, w2b, V * F1);
  k_retrieval_partial<<<32 * MSPLIT, 256, 0, stream>>>(qn, cnb, eT, pacc, pmax, psum);
  k_combine<<<(BS * E) / 256, 256, 0, stream>>>(pacc, pmax, psum, retr);
  k_fc1<<<(BS / 16) * (F1 / 16) / 8, 256, 0, stream>>>(retr, w1b, fc1_b, hbuf);
  k_fc2<<<32 * (V / 128), 256, 0, stream>>>(hbuf, w2b, fc2_b, (float*)d_out);
}
